// Bottleneck_60464549593402
// MI455X (gfx1250) — 8-bit matrix operand order repaired and checked by reading, compiled, not run on hardware
//
#include <hip/hip_runtime.h>
#include <hip/hip_bf16.h>


#define NB   64
#define HW   28
#define S    784
#define MPIX 50176
#define C1   512
#define C2   128
#define C4   512
#define EPSF 1e-5f

typedef __attribute__((ext_vector_type(8))) int v8i;

static __device__ inline v8i v8i_zero() {
    v8i z = {0, 0, 0, 0, 0, 0, 0, 0};
    return z;
}

static __device__ inline v8i wmma_iu8(v8i a, v8i b, v8i c) {
    return __builtin_amdgcn_wmma_i32_16x16x64_iu8(true, a, true, b, c, false, false);
}

static __device__ inline void async_copy16(unsigned lds_addr, const void* gptr) {
    asm volatile("global_load_async_to_lds_b128 %0, %1, off"
                 :: "v"(lds_addr), "v"(gptr) : "memory");
}
static __device__ inline void wait_async0() {
    asm volatile("s_wait_asynccnt 0" ::: "memory");
}

static __device__ inline v8i load_a(const signed char* p) {
    union { v8i v; long l[4]; } a;
    a.l[0] = *(const long*)(p);
    a.l[1] = *(const long*)(p + 16);
    a.l[2] = *(const long*)(p + 32);
    a.l[3] = *(const long*)(p + 48);
    return a.v;
}

static __device__ inline v8i load_b_lds(const signed char* p) {
    union { v8i v; int4 q[2]; } b;
    b.q[0] = *(const int4*)(p);
    b.q[1] = *(const int4*)(p + 16);
    return b.v;
}


__global__ void zero_kernel(float* p, int n) {
    int i = blockIdx.x * blockDim.x + threadIdx.x;
    if (i < n) p[i] = 0.f;
}

__global__ __launch_bounds__(256) void bn_stats_kernel(
        const float* __restrict__ x, int C,
        float* __restrict__ mean, float* __restrict__ rstd) {
    const int c = blockIdx.x;
    float s1 = 0.f, s2 = 0.f;
    for (int n = 0; n < NB; ++n) {
        const float* p = x + ((size_t)n * C + c) * S;
        for (int i = threadIdx.x; i < S; i += 256) {
            float v = p[i];
            s1 += v;
            s2 += v * v;
        }
    }
    __shared__ float r1[256], r2[256];
    r1[threadIdx.x] = s1;
    r2[threadIdx.x] = s2;
    __syncthreads();
    for (int off = 128; off > 0; off >>= 1) {
        if ((int)threadIdx.x < off) {
            r1[threadIdx.x] += r1[threadIdx.x + off];
            r2[threadIdx.x] += r2[threadIdx.x + off];
        }
        __syncthreads();
    }
    if (threadIdx.x == 0) {
        float inv = 1.f / ((float)NB * (float)S);
        float m = r1[0] * inv;
        float var = r2[0] * inv - m * m;
        mean[c] = m;
        rstd[c] = rsqrtf(var + EPSF);
    }
}

__global__ __launch_bounds__(256) void abssum_kernel(
        const float* __restrict__ x, int C,
        const float* __restrict__ mean, const float* __restrict__ rstd,
        const float* __restrict__ g, const float* __restrict__ b,
        float* __restrict__ out) {
    size_t total = (size_t)NB * (size_t)C * S;
    float acc = 0.f;
    for (size_t i = (size_t)blockIdx.x * blockDim.x + threadIdx.x; i < total;
         i += (size_t)gridDim.x * blockDim.x) {
        int c = (int)((i / S) % C);
        float xn = (x[i] - mean[c]) * rstd[c] * g[c] + b[c];
        acc += fabsf(xn);
    }
    __shared__ float r[256];
    r[threadIdx.x] = acc;
    __syncthreads();
    for (int off = 128; off > 0; off >>= 1) {
        if ((int)threadIdx.x < off) r[threadIdx.x] += r[threadIdx.x + off];
        __syncthreads();
    }
    if (threadIdx.x == 0) atomicAdd(out, r[0]);
}

__global__ __launch_bounds__(256) void quantize_kernel(
        const float* __restrict__ x, int C,
        const float* __restrict__ mean, const float* __restrict__ rstd,
        const float* __restrict__ g, const float* __restrict__ b,
        const float* __restrict__ abssum, signed char* __restrict__ q) {
    size_t total = (size_t)NB * (size_t)C * S;
    float delta = 0.7f * abssum[0] / (float)total;
    for (size_t i = (size_t)blockIdx.x * blockDim.x + threadIdx.x; i < total;
         i += (size_t)gridDim.x * blockDim.x) {
        size_t s = i % S;
        size_t c = (i / S) % C;
        size_t n = i / ((size_t)S * C);
        float xn = (x[i] - mean[c]) * rstd[c] * g[c] + b[c];
        signed char qq = (signed char)((xn > delta) ? 1 : ((xn < -delta) ? -1 : 0));
        q[(n * S + s) * C + c] = qq;
    }
}

__global__ __launch_bounds__(256) void walpha_kernel(
        const float* __restrict__ w, int Kw, float* __restrict__ alpha) {
    int o = blockIdx.x;
    float s = 0.f;
    for (int i = threadIdx.x; i < Kw; i += 256) s += fabsf(w[(size_t)o * Kw + i]);
    __shared__ float r[256];
    r[threadIdx.x] = s;
    __syncthreads();
    for (int off = 128; off > 0; off >>= 1) {
        if ((int)threadIdx.x < off) r[threadIdx.x] += r[threadIdx.x + off];
        __syncthreads();
    }
    if (threadIdx.x == 0) alpha[o] = r[0] / (float)Kw;
}

__global__ __launch_bounds__(256) void wpack_kernel(
        const float* __restrict__ w, int Cin, int Taps, int Ktot, int Nout,
        signed char* __restrict__ bp) {
    size_t total = (size_t)Ktot * Nout;
    int ntiles = Nout >> 4;
    for (size_t idx = (size_t)blockIdx.x * blockDim.x + threadIdx.x; idx < total;
         idx += (size_t)gridDim.x * blockDim.x) {
        int byt = (int)(idx & 3);
        int v = (int)((idx >> 2) & 7);
        int L = (int)((idx >> 5) & 31);
        size_t t = idx >> 10;
        int nt = (int)(t % ntiles);
        int kc = (int)(t / ntiles);
        int K = kc * 64 + ((v >> 1) << 4) + ((L >> 4) << 3) + ((v & 1) << 2) + byt;
        int n = nt * 16 + (L & 15);
        int tap = K / Cin;
        int c = K % Cin;
        float wv = w[((size_t)n * Cin + c) * (size_t)Taps + tap];
        bp[idx] = (signed char)((wv > 0.f) ? 1 : ((wv < 0.f) ? -1 : 0));
    }
}


__global__ __launch_bounds__(256) void gemm1_kernel(
        const signed char* __restrict__ act, const signed char* __restrict__ wp,
        const float* __restrict__ alpha, float* __restrict__ hout) {
    __shared__ signed char bsh[65536];
    for (int i = threadIdx.x * 16; i < 65536; i += 4096)
        async_copy16((unsigned)(uintptr_t)(bsh + i), wp + i);
    wait_async0();
    __syncthreads();

    int wave = threadIdx.x >> 5, lane = threadIdx.x & 31;
    int row = lane & 15, hi = lane >> 4;
    int mt = blockIdx.x * 8 + wave;
    const signed char* abase = act + (size_t)(mt * 16 + row) * C1 + (hi << 3);

    v8i acc[8];
#pragma unroll
    for (int nt = 0; nt < 8; ++nt) acc[nt] = v8i_zero();

    for (int kc = 0; kc < 8; ++kc) {
        if (kc < 7) __builtin_prefetch(abase + (kc + 1) * 64, 0, 0);
        v8i a = load_a(abase + kc * 64);
        const signed char* bb = bsh + kc * 8192 + lane * 32;
#pragma unroll
        for (int nt = 0; nt < 8; ++nt)
            acc[nt] = wmma_iu8(a, load_b_lds(bb + nt * 1024), acc[nt]);
    }

    int p0 = mt * 16 + (hi << 3);
    int ni = p0 / S, s0 = p0 % S;
#pragma unroll
    for (int nt = 0; nt < 8; ++nt) {
        int c = nt * 16 + row;
        float al = alpha[c];
        float* o = hout + ((size_t)ni * C2 + c) * S + s0;
        *(float4*)(o) = make_float4(al * (float)acc[nt][0], al * (float)acc[nt][1],
                                    al * (float)acc[nt][2], al * (float)acc[nt][3]);
        *(float4*)(o + 4) = make_float4(al * (float)acc[nt][4], al * (float)acc[nt][5],
                                        al * (float)acc[nt][6], al * (float)acc[nt][7]);
    }
}

__global__ __launch_bounds__(256) void gemm2_kernel(
        const signed char* __restrict__ act, const signed char* __restrict__ wp,
        const float* __restrict__ alpha, float* __restrict__ hout) {
    extern __shared__ signed char bsh[];
    for (int i = threadIdx.x * 16; i < 18 * 8192; i += 4096)
        async_copy16((unsigned)(uintptr_t)(bsh + i), wp + i);
    wait_async0();
    __syncthreads();

    int wave = threadIdx.x >> 5, lane = threadIdx.x & 31;
    int row = lane & 15, hi = lane >> 4;
    int mt = blockIdx.x * 8 + wave;
    int m = mt * 16 + row;
    int ni = m / S, s = m % S, y = s / HW, x = s % HW;
    int k0off = hi << 3;

    v8i acc[8];
#pragma unroll
    for (int nt = 0; nt < 8; ++nt) acc[nt] = v8i_zero();

    for (int tap = 0; tap < 9; ++tap) {
        int yy = y + tap / 3 - 1, xx = x + tap % 3 - 1;
        bool valid = ((unsigned)yy < (unsigned)HW) && ((unsigned)xx < (unsigned)HW);
        long pix = (long)ni * S + (long)yy * HW + xx;
        const signed char* ap = act + pix * C2 + k0off;
        if (valid) __builtin_prefetch(ap, 0, 0);

#pragma unroll
        for (int ch = 0; ch < 2; ++ch) {
            v8i a = valid ? load_a(ap + ch * 64) : v8i_zero();
            const signed char* bb = bsh + (tap * 2 + ch) * 8192 + lane * 32;
#pragma unroll
            for (int nt = 0; nt < 8; ++nt)
                acc[nt] = wmma_iu8(a, load_b_lds(bb + nt * 1024), acc[nt]);
        }
    }

    int p0 = mt * 16 + (hi << 3);
    int n2 = p0 / S, s0 = p0 % S;
#pragma unroll
    for (int nt = 0; nt < 8; ++nt) {
        int c = nt * 16 + row;
        float al = alpha[c];
        float* o = hout + ((size_t)n2 * C2 + c) * S + s0;
        *(float4*)(o) = make_float4(al * (float)acc[nt][0], al * (float)acc[nt][1],
                                    al * (float)acc[nt][2], al * (float)acc[nt][3]);
        *(float4*)(o + 4) = make_float4(al * (float)acc[nt][4], al * (float)acc[nt][5],
                                        al * (float)acc[nt][6], al * (float)acc[nt][7]);
    }
}

__global__ __launch_bounds__(256) void gemm3_kernel(
        const signed char* __restrict__ act, const signed char* __restrict__ wp,
        const float* __restrict__ alpha, const float* __restrict__ xin,
        float* __restrict__ out) {
    __shared__ signed char bsh[65536];
    for (int i = threadIdx.x * 16; i < 65536; i += 4096)
        async_copy16((unsigned)(uintptr_t)(bsh + i), wp + i);
    wait_async0();
    __syncthreads();

    int wave = threadIdx.x >> 5, lane = threadIdx.x & 31;
    int row = lane & 15, hi = lane >> 4;
    int mt = blockIdx.x * 8 + wave;
    int ng = blockIdx.y;
    const signed char* abase = act + (size_t)(mt * 16 + row) * C2 + (hi << 3);

    v8i acc[8];
#pragma unroll
    for (int j = 0; j < 8; ++j) acc[j] = v8i_zero();

#pragma unroll
    for (int kc = 0; kc < 2; ++kc) {
        v8i a = load_a(abase + kc * 64);
#pragma unroll
        for (int j = 0; j < 8; ++j) {
            int nt = ng * 8 + j;
            const signed char* bb = bsh + (kc * 32 + nt) * 1024 + lane * 32;
            acc[j] = wmma_iu8(a, load_b_lds(bb), acc[j]);
        }
    }

    int p0 = mt * 16 + (hi << 3);
    int ni = p0 / S, s0 = p0 % S;
#pragma unroll
    for (int j = 0; j < 8; ++j) {
        int c = (ng * 8 + j) * 16 + row;
        float al = alpha[c];
        size_t off = ((size_t)ni * C4 + c) * S + s0;
        const float* xi = xin + off;
        float* o = out + off;
        *(float4*)(o) = make_float4(xi[0] + al * (float)acc[j][0],
                                    xi[1] + al * (float)acc[j][1],
                                    xi[2] + al * (float)acc[j][2],
                                    xi[3] + al * (float)acc[j][3]);
        *(float4*)(o + 4) = make_float4(xi[4] + al * (float)acc[j][4],
                                        xi[5] + al * (float)acc[j][5],
                                        xi[6] + al * (float)acc[j][6],
                                        xi[7] + al * (float)acc[j][7]);
    }
}


extern "C" void kernel_launch(void* const* d_in, const int* in_sizes, int n_in,
                              void* d_out, int out_size, void* d_ws, size_t ws_size,
                              hipStream_t stream) {
    (void)in_sizes; (void)n_in; (void)out_size; (void)ws_size;

    const float* x  = (const float*)d_in[0];
    const float* g1 = (const float*)d_in[1];
    const float* b1 = (const float*)d_in[2];
    const float* w1 = (const float*)d_in[3];
    const float* g2 = (const float*)d_in[4];
    const float* b2 = (const float*)d_in[5];
    const float* w2 = (const float*)d_in[6];
    const float* g3 = (const float*)d_in[7];
    const float* b3 = (const float*)d_in[8];
    const float* w3 = (const float*)d_in[9];
    float* out = (float*)d_out;
    char* ws = (char*)d_ws;

    size_t o = 0;
    signed char* actA = (signed char*)(ws + o); o += (size_t)MPIX * C1;
    signed char* actB = (signed char*)(ws + o); o += (size_t)MPIX * C2;
    signed char* actC = (signed char*)(ws + o); o += (size_t)MPIX * C2;
    float* h1 = (float*)(ws + o); o += (size_t)MPIX * C2 * 4;
    float* h2 = (float*)(ws + o); o += (size_t)MPIX * C2 * 4;
    signed char* w1p = (signed char*)(ws + o); o += 8 * 8 * 1024;
    signed char* w2p = (signed char*)(ws + o); o += 18 * 8 * 1024;
    signed char* w3p = (signed char*)(ws + o); o += 2 * 32 * 1024;
    float* alpha1 = (float*)(ws + o); o += 2048;
    float* alpha2 = (float*)(ws + o); o += 2048;
    float* alpha3 = (float*)(ws + o); o += 2048;
    float* mean1 = (float*)(ws + o); o += 2048;
    float* rstd1 = (float*)(ws + o); o += 2048;
    float* mean2 = (float*)(ws + o); o += 2048;
    float* rstd2 = (float*)(ws + o); o += 2048;
    float* mean3 = (float*)(ws + o); o += 2048;
    float* rstd3 = (float*)(ws + o); o += 2048;
    float* scal = (float*)(ws + o); o += 256;

    zero_kernel<<<1, 32, 0, stream>>>(scal, 3);

    bn_stats_kernel<<<C1, 256, 0, stream>>>(x, C1, mean1, rstd1);
    abssum_kernel<<<1024, 256, 0, stream>>>(x, C1, mean1, rstd1, g1, b1, scal + 0);
    quantize_kernel<<<2048, 256, 0, stream>>>(x, C1, mean1, rstd1, g1, b1, scal + 0, actA);
    walpha_kernel<<<C2, 256, 0, stream>>>(w1, C1, alpha1);
    wpack_kernel<<<64, 256, 0, stream>>>(w1, C1, 1, C1, C2, w1p);
    gemm1_kernel<<<MPIX / (16 * 8), 256, 0, stream>>>(actA, w1p, alpha1, h1);

    bn_stats_kernel<<<C2, 256, 0, stream>>>(h1, C2, mean2, rstd2);
    abssum_kernel<<<1024, 256, 0, stream>>>(h1, C2, mean2, rstd2, g2, b2, scal + 1);
    quantize_kernel<<<2048, 256, 0, stream>>>(h1, C2, mean2, rstd2, g2, b2, scal + 1, actB);
    walpha_kernel<<<C2, 256, 0, stream>>>(w2, C2 * 9, alpha2);
    wpack_kernel<<<144, 256, 0, stream>>>(w2, C2, 9, C2 * 9, C2, w2p);
    gemm2_kernel<<<MPIX / (16 * 8), 256, 18 * 8192, stream>>>(actB, w2p, alpha2, h2);

    bn_stats_kernel<<<C2, 256, 0, stream>>>(h2, C2, mean3, rstd3);
    abssum_kernel<<<1024, 256, 0, stream>>>(h2, C2, mean3, rstd3, g3, b3, scal + 2);
    quantize_kernel<<<2048, 256, 0, stream>>>(h2, C2, mean3, rstd3, g3, b3, scal + 2, actC);
    walpha_kernel<<<C4, 256, 0, stream>>>(w3, C2, alpha3);
    wpack_kernel<<<64, 256, 0, stream>>>(w3, C2, 1, C2, C4, w3p);
    gemm3_kernel<<<dim3(MPIX / (16 * 8), 4), 256, 0, stream>>>(actC, w3p, alpha3, x, out);
}
